// se3ACN_49709951484149
// MI455X (gfx1250) — matrix operand order repaired and checked by reading, not compiled, not run on hardware
//
#include <hip/hip_runtime.h>
#include <hip/hip_bf16.h>
#include <stdint.h>

typedef __attribute__((ext_vector_type(16))) _Float16 v16h;
typedef __attribute__((ext_vector_type(8)))  float    v8f;

#define BB   8
#define NN   64
#define EMBD 32
#define CDIM 24

// ---- WMMA fragment index helpers (CDNA5 ISA 7.12.2, wave32) ----
static __device__ __forceinline__ int a_koff(int lane, int idx) {
  return ((idx >> 3) << 4) + ((lane >> 4) << 3) + (idx & 7);
}
static __device__ __forceinline__ int b_koff(int lane, int idx) {
  return ((idx >> 3) << 4) + ((lane >> 4) << 3) + (idx & 7);
}
static __device__ __forceinline__ int cd_row(int lane, int r) {
  return ((lane >> 4) << 3) + r;
}

static __device__ __forceinline__ float softplus5(float x) {
  float t = 5.f * x;
  return (t > 20.f) ? x : log1pf(expf(t)) * 0.2f;
}

// ---------------- K1: geometry / basis / SH / embedding gather ----------------
// grid = B*N blocks (one per (b,i)), 64 threads (one per j)
__global__ void geom_kernel(const float* __restrict__ xyz, const int* __restrict__ feat,
                            const float* __restrict__ emb,
                            float* __restrict__ f32o, float* __restrict__ masko,
                            float* __restrict__ Yo, float* __restrict__ baso,
                            float* __restrict__ invo) {
  __shared__ float cnt[NN];
  int bi = blockIdx.x;        // b*64 + i
  int j  = threadIdx.x;
  int b  = bi >> 6;
  const float* pi = xyz + (size_t)bi * 3;
  const float* pj = xyz + (size_t)(b * NN + j) * 3;
  float rx = pj[0] - pi[0], ry = pj[1] - pi[1], rz = pj[2] - pi[2];
  float d = sqrtf(rx * rx + ry * ry + rz * rz + 1e-8f);
  float m = ((d < 2.0f) && (j != (bi & 63))) ? 1.f : 0.f;
  int p = bi * NN + j;
  masko[p] = m;
  #pragma unroll
  for (int q = 0; q < 3; ++q) {         // radii = {0,1,2}, step = 1
    float u = fabsf(d - (float)q);
    float c = cosf(1.57079632679f * u);
    baso[p * 3 + q] = (u < 1.f) ? c * c : 0.f;
  }
  float ux = rx / d, uy = ry / d, uz = rz / d;
  float* Yp = Yo + (size_t)p * 8;
  Yp[0] = 0.48860251190f * uy;
  Yp[1] = 0.48860251190f * uz;
  Yp[2] = 0.48860251190f * ux;
  Yp[3] = 1.09254843059f * ux * uy;
  Yp[4] = 1.09254843059f * uy * uz;
  Yp[5] = 0.31539156525f * (3.f * uz * uz - 1.f);
  Yp[6] = 1.09254843059f * ux * uz;
  Yp[7] = 0.5f * 1.09254843059f * (ux * ux - uy * uy);
  cnt[j] = m;
  __syncthreads();
  for (int s = 32; s > 0; s >>= 1) {
    if (j < s) cnt[j] += cnt[j + s];
    __syncthreads();
  }
  if (j == 0) invo[bi] = rsqrtf(fmaxf(cnt[0], 1.f));
  if (j < EMBD) f32o[(size_t)bi * EMBD + j] = emb[feat[bi] * EMBD + j];
}

// ---------------- K2: per-atom W = rW3 (x) f,  M=10240 (k*80+lu), K=32, N=512 ----------------
// grid = (640, 8), block = 128 (4 waves, one 16x16 output tile each)
__global__ void wf_kernel(const float* __restrict__ rW3, const float* __restrict__ rb3,
                          const int* __restrict__ feat, const float* __restrict__ emb,
                          _Float16* __restrict__ Wo) {
  int lane = threadIdx.x & 31;
  int wave = threadIdx.x >> 5;
  int mt = blockIdx.x;               // 0..639
  int nt = blockIdx.y * 4 + wave;    // 0..31
  v16h a, bfr;
  int mrow = mt * 16 + (lane & 15);
  int k = mrow / 80, lu = mrow % 80;
  #pragma unroll
  for (int idx = 0; idx < 16; ++idx) {
    int v = a_koff(lane, idx);
    float av = 0.f;
    if (lu < 72) {
      if (k < 100)       av = rW3[(size_t)k * 2304 + lu * 32 + v];
      else if (k == 100) av = rb3[lu * 32 + v];       // fold radial bias as row k=100
    }
    a[idx] = (_Float16)av;
  }
  int bj = nt * 16 + (lane & 15);
  int fi = feat[bj];
  #pragma unroll
  for (int idx = 0; idx < 16; ++idx)
    bfr[idx] = (_Float16)emb[fi * EMBD + b_koff(lane, idx)];
  v8f c = {};
  c = __builtin_amdgcn_wmma_f32_16x16x32_f16(false, a, false, bfr, (short)0, c, false, false);
  #pragma unroll
  for (int r = 0; r < 8; ++r) {
    int mm = mt * 16 + cd_row(lane, r);
    int nn = nt * 16 + (lane & 15);
    Wo[(size_t)nn * 10240 + mm] = (_Float16)c[r];
  }
}

// ---------------- K3: radial MLP layers 1+2, h2[32768 x 128] f16 ----------------
// grid = 512 blocks (64 pairs each = one (b,i) row), block = 128 (4 waves)
__global__ void radial_kernel(const float* __restrict__ baso,
                              const float* __restrict__ rW1, const float* __restrict__ rb1,
                              const float* __restrict__ rW2, const float* __restrict__ rb2,
                              _Float16* __restrict__ h2o) {
  __shared__ _Float16 h1s[64 * 128];
  int tid = threadIdx.x;
  int lane = tid & 31, wave = tid >> 5;
  int pair0 = blockIdx.x * 64;
  // layer 1 (K=3, pure VALU) into LDS, K-padded to 128 with zeros
  for (int e = tid; e < 64 * 128; e += 128) {
    int pp = e >> 7, n = e & 127;
    float v = 0.f;
    if (n < 100) {
      const float* bs = baso + (size_t)(pair0 + pp) * 3;
      v = softplus5(bs[0] * rW1[n] + bs[1] * rW1[100 + n] + bs[2] * rW1[200 + n] + rb1[n]);
    }
    h1s[e] = (_Float16)v;
  }
  __syncthreads();
  // layer 2 as WMMA: M=64 (4 Mtiles, one per wave), N=112 (7 Ntiles), K=128 (4 chunks)
  v16h afr[4];
  int row = wave * 16 + (lane & 15);
  #pragma unroll
  for (int ch = 0; ch < 4; ++ch)
    #pragma unroll
    for (int idx = 0; idx < 16; ++idx)
      afr[ch][idx] = h1s[row * 128 + ch * 32 + a_koff(lane, idx)];
  for (int nt = 0; nt < 7; ++nt) {
    v8f c = {};
    #pragma unroll
    for (int ch = 0; ch < 4; ++ch) {
      v16h bfr;
      int n = nt * 16 + (lane & 15);
      #pragma unroll
      for (int idx = 0; idx < 16; ++idx) {
        int kk = ch * 32 + b_koff(lane, idx);
        float bv = (kk < 100 && n < 100) ? rW2[kk * 100 + n] : 0.f;
        bfr[idx] = (_Float16)bv;
      }
      c = __builtin_amdgcn_wmma_f32_16x16x32_f16(false, afr[ch], false, bfr, (short)0, c, false, false);
    }
    #pragma unroll
    for (int r = 0; r < 8; ++r) {
      int P = wave * 16 + cd_row(lane, r);
      int n = nt * 16 + (lane & 15);
      float h = (n < 100) ? softplus5(c[r] + rb2[n]) : ((n == 100) ? 1.f : 0.f);
      h2o[(size_t)(pair0 + P) * 128 + n] = (_Float16)h;
    }
  }
  // zero the K-pad columns 112..127
  for (int e = tid; e < 64 * 16; e += 128)
    h2o[(size_t)(pair0 + (e >> 4)) * 128 + 112 + (e & 15)] = (_Float16)0.f;
}

// ---------------- K4: conv: t = h2 @ W[b,j]; masked SH accumulation over j ----------------
// grid = (8, 4): (batch, i-tile). block = 160 (5 waves; wave = N-tile of lu).
// W[b,j] panel (128x80 f16 = 20KB) is double-buffered in LDS via
// GLOBAL_LOAD_ASYNC_TO_LDS_B128 (ASYNCcnt-tracked), overlapped with the GEMM of j.
__global__ void conv_kernel(const _Float16* __restrict__ h2, const _Float16* __restrict__ W,
                            const float* __restrict__ masko, const float* __restrict__ Yo,
                            const float* __restrict__ invo, const float* __restrict__ f32in,
                            float* __restrict__ featsF, _Float16* __restrict__ feats16) {
  __shared__ _Float16 wbuf[2][10240];   // 40960 B
  __shared__ float    oacc[16 * 216];   // 13824 B
  __shared__ _Float16 ts[16 * 80];      //  2560 B
  int tid = threadIdx.x;                // 0..159
  int lane = tid & 31, wave = tid >> 5; // wave = nt (0..4)
  int b = blockIdx.x;
  int i0 = blockIdx.y * 16;
  for (int e = tid; e < 16 * 216; e += 160) oacc[e] = 0.f;

  auto stage = [&](int jj, int nb) {
    // 160 threads x 16B x 8 passes = 20480 B panel
    unsigned long long gbase =
        (unsigned long long)(uintptr_t)(W + (size_t)(b * 64 + jj) * 10240);
    unsigned lbase = (unsigned)(uintptr_t)(&wbuf[nb][0]);   // LDS offset = addr[31:0]
    #pragma unroll
    for (int pass = 0; pass < 8; ++pass) {
      unsigned loff = lbase + (unsigned)(pass * 2560 + tid * 16);
      unsigned long long ga = gbase + (unsigned long long)(pass * 2560 + tid * 16);
      asm volatile("global_load_async_to_lds_b128 %0, %1, off"
                   :: "v"(loff), "v"(ga) : "memory");
    }
  };

  stage(0, 0);
  for (int j = 0; j < 64; ++j) {
    if (j < 63) {
      stage(j + 1, (j + 1) & 1);
      asm volatile("s_wait_asynccnt 0x8" ::: "memory");  // batch j landed (in-order retire)
    } else {
      asm volatile("s_wait_asynccnt 0x0" ::: "memory");
    }
    __syncthreads();
    const _Float16* wb = &wbuf[j & 1][0];
    {
      int nt = wave;
      v8f c = {};
      #pragma unroll
      for (int ch = 0; ch < 4; ++ch) {
        v16h afr, bfr;
        int i = i0 + (lane & 15);
        const _Float16* ap = h2 + ((size_t)((b * 64 + i) * 64 + j)) * 128 + ch * 32;
        #pragma unroll
        for (int idx = 0; idx < 16; ++idx) afr[idx] = ap[a_koff(lane, idx)];
        int n = nt * 16 + (lane & 15);
        #pragma unroll
        for (int idx = 0; idx < 16; ++idx)
          bfr[idx] = wb[(ch * 32 + b_koff(lane, idx)) * 80 + n];
        c = __builtin_amdgcn_wmma_f32_16x16x32_f16(false, afr, false, bfr, (short)0, c, false, false);
      }
      #pragma unroll
      for (int r = 0; r < 8; ++r) {
        int ii = cd_row(lane, r);
        int nn = nt * 16 + (lane & 15);
        ts[ii * 80 + nn] = (_Float16)(c[r] * 0.17677669529f);   // 1/sqrt(EMB)
      }
    }
    __syncthreads();
    // masked, SH-weighted accumulation: fixed (i, channel) ownership => no atomics
    for (int e = tid; e < 16 * 216; e += 160) {
      int i = e / 216, cc = e % 216;
      int l, u, mg;
      if (cc < 24)      { l = 0; u = cc;            mg = -1; }
      else if (cc < 96) { l = 1; u = (cc - 24) / 3; mg = (cc - 24) % 3; }
      else              { l = 2; u = (cc - 96) / 5; mg = 3 + (cc - 96) % 5; }
      int p = (b * 64 + i0 + i) * 64 + j;
      float msk = masko[p];
      float y = (mg < 0) ? 0.28209479177f : Yo[(size_t)p * 8 + mg];
      oacc[e] += msk * y * (float)ts[i * 80 + l * 24 + u];
    }
    __syncthreads();
  }
  // feats = [f | conv*inv], emit f32 (for pooling) and f16 padded to 256 (next GEMM)
  for (int e = tid; e < 16 * 248; e += 160) {
    int i = e / 248, ch = e % 248;
    int bi = b * 64 + i0 + i;
    float v = (ch < 32) ? f32in[(size_t)bi * 32 + ch] : oacc[i * 216 + (ch - 32)] * invo[bi];
    featsF[(size_t)bi * 248 + ch] = v;
    feats16[(size_t)bi * 256 + ch] = (_Float16)v;
  }
  for (int e = tid; e < 16 * 8; e += 160)
    feats16[(size_t)(b * 64 + i0 + (e >> 3)) * 256 + 248 + (e & 7)] = (_Float16)0.f;
}

// ---------------- K5: x2 = feats @ resW + resb   (512 x 256 x 256 WMMA) ----------------
// grid = (32, 4), block = 128 (4 waves); 16 Ntiles total
__global__ void res_gemm_kernel(const _Float16* __restrict__ feats16,
                                const float* __restrict__ resW, const float* __restrict__ resb,
                                float* __restrict__ x2o) {
  int lane = threadIdx.x & 31, wave = threadIdx.x >> 5;
  int mt = blockIdx.x;
  int nt = blockIdx.y * 4 + wave;
  v8f c = {};
  int arow = mt * 16 + (lane & 15);
  int n = nt * 16 + (lane & 15);
  for (int ch = 0; ch < 8; ++ch) {
    v16h afr, bfr;
    const _Float16* ap = feats16 + (size_t)arow * 256 + ch * 32;
    #pragma unroll
    for (int idx = 0; idx < 16; ++idx) afr[idx] = ap[a_koff(lane, idx)];
    #pragma unroll
    for (int idx = 0; idx < 16; ++idx) {
      int kk = ch * 32 + b_koff(lane, idx);
      float bv = (kk < 248 && n < 248) ? resW[(size_t)kk * 248 + n] : 0.f;
      bfr[idx] = (_Float16)bv;
    }
    c = __builtin_amdgcn_wmma_f32_16x16x32_f16(false, afr, false, bfr, (short)0, c, false, false);
  }
  #pragma unroll
  for (int r = 0; r < 8; ++r) {
    int mm = mt * 16 + cd_row(lane, r);
    int nn = nt * 16 + (lane & 15);
    if (nn < 248) x2o[(size_t)mm * 248 + nn] = c[r] + resb[nn];
  }
}

// ---------------- K6: batchnorm + relu + concat + mean-pool + head MLP ----------------
// single block, 256 threads
__global__ void tail_kernel(const float* __restrict__ featsF, const float* __restrict__ x2,
                            const float* __restrict__ gam, const float* __restrict__ bet,
                            const float* __restrict__ cW, const float* __restrict__ cb,
                            const float* __restrict__ cg, const float* __restrict__ cbt,
                            const float* __restrict__ oW, const float* __restrict__ ob,
                            float* __restrict__ out) {
  __shared__ float pooled[8 * 496];
  __shared__ float hsh[8 * 128];
  int tid = threadIdx.x;
  if (tid < 248) {               // per-column BN stats over 512 rows (population var)
    float s = 0.f, ss = 0.f;
    for (int r = 0; r < 512; ++r) { float v = x2[(size_t)r * 248 + tid]; s += v; ss += v * v; }
    float mu = s * (1.f / 512.f);
    float var = ss * (1.f / 512.f) - mu * mu;
    float sc = gam[tid] * rsqrtf(var + 1e-5f);
    float sh = bet[tid] - mu * sc;
    for (int b = 0; b < 8; ++b) {
      float a0 = 0.f, a1 = 0.f;
      for (int i = 0; i < 64; ++i) {
        int r = b * 64 + i;
        a0 += featsF[(size_t)r * 248 + tid];
        float v = x2[(size_t)r * 248 + tid] * sc + sh;
        a1 += fmaxf(v, 0.f);
      }
      pooled[b * 496 + tid] = a0 * (1.f / 64.f);
      pooled[b * 496 + 248 + tid] = a1 * (1.f / 64.f);
    }
  }
  __syncthreads();
  for (int e = tid; e < 8 * 128; e += 256) {     // pooled @ cW + cb, leaky
    int b = e >> 7, cc2 = e & 127;
    float acc = cb[cc2];
    for (int k = 0; k < 496; ++k) acc += pooled[b * 496 + k] * cW[k * 128 + cc2];
    hsh[e] = fmaxf(acc, 0.01f * acc);
  }
  __syncthreads();
  if (tid < 128) {               // BN over 8 rows, leaky
    float s = 0.f, ss = 0.f;
    for (int b = 0; b < 8; ++b) { float v = hsh[b * 128 + tid]; s += v; ss += v * v; }
    float mu = s * 0.125f, var = ss * 0.125f - mu * mu;
    float sc = cg[tid] * rsqrtf(var + 1e-5f);
    float sh = cbt[tid] - mu * sc;
    for (int b = 0; b < 8; ++b) {
      float v = hsh[b * 128 + tid] * sc + sh;
      hsh[b * 128 + tid] = fmaxf(v, 0.01f * v);
    }
  }
  __syncthreads();
  if (tid < 8) {                 // sigmoid(h @ oW + ob)
    float acc = ob[0];
    for (int k = 0; k < 128; ++k) acc += hsh[tid * 128 + k] * oW[k];
    out[tid] = 1.f / (1.f + expf(-acc));
  }
}

extern "C" void kernel_launch(void* const* d_in, const int* in_sizes, int n_in,
                              void* d_out, int out_size, void* d_ws, size_t ws_size,
                              hipStream_t stream) {
  (void)in_sizes; (void)n_in; (void)out_size; (void)ws_size;
  const float* xyz  = (const float*)d_in[0];
  const int*   feat = (const int*)d_in[1];
  const float* emb  = (const float*)d_in[2];
  const float* rW1  = (const float*)d_in[3];
  const float* rb1  = (const float*)d_in[4];
  const float* rW2  = (const float*)d_in[5];
  const float* rb2  = (const float*)d_in[6];
  const float* rW3  = (const float*)d_in[7];
  const float* rb3  = (const float*)d_in[8];
  const float* resW = (const float*)d_in[9];
  const float* resb = (const float*)d_in[10];
  const float* rg   = (const float*)d_in[11];
  const float* rbta = (const float*)d_in[12];
  const float* cW   = (const float*)d_in[13];
  const float* cb   = (const float*)d_in[14];
  const float* cg   = (const float*)d_in[15];
  const float* cbt  = (const float*)d_in[16];
  const float* oW   = (const float*)d_in[17];
  const float* ob   = (const float*)d_in[18];
  float* out = (float*)d_out;

  char* ws = (char*)d_ws;
  size_t off = 0;
  auto alloc = [&](size_t bytes) {
    size_t o = off;
    off = (off + bytes + 255) & ~(size_t)255;
    return o;
  };
  float*    f32b   = (float*)(ws + alloc((size_t)512 * 32 * 4));
  float*    maskb  = (float*)(ws + alloc((size_t)32768 * 4));
  float*    Yb     = (float*)(ws + alloc((size_t)32768 * 8 * 4));
  float*    basb   = (float*)(ws + alloc((size_t)32768 * 3 * 4));
  float*    invb   = (float*)(ws + alloc((size_t)512 * 4));
  _Float16* h2b    = (_Float16*)(ws + alloc((size_t)32768 * 128 * 2));
  _Float16* Wb     = (_Float16*)(ws + alloc((size_t)512 * 10240 * 2));
  float*    featsb = (float*)(ws + alloc((size_t)512 * 248 * 4));
  _Float16* f16b   = (_Float16*)(ws + alloc((size_t)512 * 256 * 2));
  float*    x2b    = (float*)(ws + alloc((size_t)512 * 248 * 4));

  geom_kernel  <<<512, 64, 0, stream>>>(xyz, feat, emb, f32b, maskb, Yb, basb, invb);
  wf_kernel    <<<dim3(640, 8), 128, 0, stream>>>(rW3, rb3, feat, emb, Wb);
  radial_kernel<<<512, 128, 0, stream>>>(basb, rW1, rb1, rW2, rb2, h2b);
  conv_kernel  <<<dim3(8, 4), 160, 0, stream>>>(h2b, Wb, maskb, Yb, invb, f32b, featsb, f16b);
  res_gemm_kernel<<<dim3(32, 4), 128, 0, stream>>>(f16b, resW, resb, x2b);
  tail_kernel  <<<1, 256, 0, stream>>>(featsb, x2b, rg, rbta, cW, cb, cg, cbt, oW, ob, out);
}
